// NILMFormerSeq2Seq_54099408060580
// MI455X (gfx1250) — matrix operand order repaired and checked by reading, not compiled, not run on hardware
//
#include <hip/hip_runtime.h>

// ---------------- model constants ----------------
#define Bc   16
#define Lc   1024
#define Dc   256
#define DEc  192
#define Hc   8
#define HDc  32
#define FFc  1024
#define Sc   1025          // L + 1 stats token
#define SPc  1040          // padded to 65 * 16
#define Rc   (Bc * SPc)    // 16640 rows = 130 * 128
#define WAVES 8
#define KCH  128           // GEMM K-chunk staged in LDS

typedef __attribute__((ext_vector_type(16))) __bf16 v16bf;
typedef __attribute__((ext_vector_type(8)))  __bf16 v8bf;
typedef __attribute__((ext_vector_type(8)))  float  v8f;

#define WMMA_BF16(a, b, c) \
  __builtin_amdgcn_wmma_f32_16x16x32_bf16(false, (a), false, (b), (short)0, (c), false, false)

// ---------------- fragment helpers (16x16x32 bf16; this code's contiguous k order, same in A and B) ----------------
// A (16x32, MxK): lane = m + 16*g ; element e -> K = e + 16*g
// => per lane: two contiguous 8-element chunks at K = 16g and K = 16g+8.
__device__ __forceinline__ v16bf a_frag_f32(const float* row, int g) {
  v16bf a;
  const float* p0 = row + 16 * g;
  const float* p1 = row + 16 * g + 8;
#pragma unroll
  for (int i = 0; i < 8; ++i) { a[i] = (__bf16)p0[i]; a[8 + i] = (__bf16)p1[i]; }
  return a;
}
__device__ __forceinline__ v16bf a_frag_bf16(const __bf16* row, int g) {
  v8bf lo = *(const v8bf*)(row + 16 * g);
  v8bf hi = *(const v8bf*)(row + 16 * g + 8);
  return __builtin_shufflevector(lo, hi, 0,1,2,3,4,5,6,7,8,9,10,11,12,13,14,15);
}
__device__ __forceinline__ v16bf a_frag_zero() {
  v16bf a;
#pragma unroll
  for (int i = 0; i < 16; ++i) a[i] = (__bf16)0.0f;
  return a;
}
// B (32x16, KxN) from packed [n][k] bf16 (k contiguous):
// lane = n + 16*gk ; element e -> K = e + 16*gk  => one contiguous 32B load.
__device__ __forceinline__ v16bf b_frag(const __bf16* pack, size_t ldk, int n0, int k0, int lane) {
  return *(const v16bf*)(pack + (size_t)(n0 + (lane & 15)) * ldk + k0 + 16 * (lane >> 4));
}

__device__ __forceinline__ float gelu_exact(float x) {
  return 0.5f * x * (1.0f + erff(x * 0.70710678118654752f));
}

// ---------------- packing kernels ----------------
// W[k][n] (fp32) -> P[n][k] (bf16)
__global__ void k_pack(const float* __restrict__ W, __bf16* __restrict__ P, int K, int N) {
  int idx = blockIdx.x * blockDim.x + threadIdx.x;
  if (idx >= K * N) return;
  int k = idx % K, n = idx / K;
  P[(size_t)n * K + k] = (__bf16)W[(size_t)k * N + n];
}
// conv w[co][ci][3] -> P[t][co][ci]
__global__ void k_pack_conv(const float* __restrict__ W, __bf16* __restrict__ P) {
  int idx = blockIdx.x * blockDim.x + threadIdx.x;
  if (idx >= 3 * DEc * DEc) return;
  int ci = idx % DEc, co = (idx / DEc) % DEc, t = idx / (DEc * DEc);
  P[idx] = (__bf16)W[((size_t)co * DEc + ci) * 3 + t];
}

// ---------------- stage 1: instance norm + stats tokens ----------------
__global__ void k_instnorm(const float* __restrict__ x, float* __restrict__ mn,
                           const float* __restrict__ ps1w, const float* __restrict__ ps1b,
                           const float* __restrict__ ps2w, const float* __restrict__ ps2b,
                           float* __restrict__ stok, float* __restrict__ so) {
  int b = blockIdx.x, t = threadIdx.x;
  __shared__ float sh[256];
  const float* xb = x + (size_t)b * Lc;
  float s = 0.f;
  for (int i = t; i < Lc; i += 256) s += xb[i];
  sh[t] = s; __syncthreads();
  for (int o = 128; o; o >>= 1) { if (t < o) sh[t] += sh[t + o]; __syncthreads(); }
  float mean = sh[0] * (1.0f / Lc); __syncthreads();
  float s2 = 0.f;
  for (int i = t; i < Lc; i += 256) { float d = xb[i] - mean; s2 += d * d; }
  sh[t] = s2; __syncthreads();
  for (int o = 128; o; o >>= 1) { if (t < o) sh[t] += sh[t + o]; __syncthreads(); }
  float stdv = sqrtf(sh[0] * (1.0f / Lc) + 1e-6f); __syncthreads();
  float inv = 1.0f / stdv;
  for (int i = t; i < Lc; i += 256) mn[(size_t)b * Lc + i] = (xb[i] - mean) * inv;
  float sv = mean * ps1w[t] + stdv * ps1w[256 + t] + ps1b[t];
  stok[(size_t)b * Dc + t] = sv;
  sh[t] = sv * ps2w[t * 2 + 0]; __syncthreads();
  for (int o = 128; o; o >>= 1) { if (t < o) sh[t] += sh[t + o]; __syncthreads(); }
  if (t == 0) so[b * 2 + 0] = sh[0] + ps2b[0];
  __syncthreads();
  sh[t] = sv * ps2w[t * 2 + 1]; __syncthreads();
  for (int o = 128; o; o >>= 1) { if (t < o) sh[t] += sh[t + o]; __syncthreads(); }
  if (t == 0) so[b * 2 + 1] = sh[0] + ps2b[1];
}

// ---------------- conv block 1 (cin=1, dil=1): direct ----------------
__global__ void k_embed0(const float* __restrict__ mn, const float* __restrict__ w,
                         const float* __restrict__ bias,
                         const float* __restrict__ bng, const float* __restrict__ bnb,
                         const float* __restrict__ bnrm, const float* __restrict__ bnrv,
                         float* __restrict__ h0) {
  int idx = blockIdx.x * blockDim.x + threadIdx.x;
  if (idx >= Bc * Lc * DEc) return;
  int co = idx % DEc, l = (idx / DEc) % Lc, b = idx / (DEc * Lc);
  const float* xm = mn + (size_t)b * Lc;
  float acc = bias[co];
#pragma unroll
  for (int t = 0; t < 3; ++t) {
    int ll = l + t - 1;
    if (ll >= 0 && ll < Lc) acc += w[co * 3 + t] * xm[ll];
  }
  float y = gelu_exact(acc);
  y = (y - bnrm[co]) * rsqrtf(bnrv[co] + 1e-5f) * bng[co] + bnb[co];
  h0[idx] = xm[l] + y;
}

// ---------------- conv blocks 2..4 (192->192, K=3, dilated) : WMMA ----------------
__global__ void k_conv(const float* __restrict__ hin, const __bf16* __restrict__ wp /*[3][192][192]*/,
                       const float* __restrict__ bias,
                       const float* __restrict__ bng, const float* __restrict__ bnb,
                       const float* __restrict__ bnrm, const float* __restrict__ bnrv,
                       float* __restrict__ hout, int dil) {
  int wave = threadIdx.x >> 5, lane = threadIdx.x & 31;
  int tile = blockIdx.x * WAVES + wave;
  const int NT = DEc / 16;                    // 12
  if (tile >= Bc * (Lc / 16) * NT) return;
  int nt = tile % NT, lt = (tile / NT) % (Lc / 16), b = tile / (NT * (Lc / 16));
  int m = lane & 15, g = lane >> 4;
  int l = lt * 16 + m;
  v8f acc = {};
#pragma unroll
  for (int t = 0; t < 3; ++t) {
    int lr_ = l + (t - 1) * dil;
    bool valid = (lr_ >= 0) && (lr_ < Lc);
    const __bf16* bt = wp + (size_t)t * DEc * DEc;
    for (int k0 = 0; k0 < DEc; k0 += 32) {
      v16bf a;
      if (valid) a = a_frag_f32(hin + ((size_t)b * Lc + lr_) * DEc + k0, g);
      else       a = a_frag_zero();
      v16bf bb = b_frag(bt, DEc, nt * 16, k0, lane);
      acc = WMMA_BF16(a, bb, acc);
    }
  }
  int n = nt * 16 + (lane & 15);
  float bv = bias[n];
  float bnr = rsqrtf(bnrv[n] + 1e-5f);
#pragma unroll
  for (int r = 0; r < 8; ++r) {
    int lo = lt * 16 + r + 8 * g;
    float y = gelu_exact(acc[r] + bv);
    y = (y - bnrm[n]) * bnr * bng[n] + bnb[n];
    size_t off = ((size_t)b * Lc + lo) * DEc + n;
    hout[off] = hin[off] + y;
  }
}

// ---------------- assemble hseq ----------------
__global__ void k_build_hseq(const float* __restrict__ h, const float* __restrict__ projw,
                             const float* __restrict__ projb, const float* __restrict__ stok,
                             float* __restrict__ hseq) {
  int idx = blockIdx.x * blockDim.x + threadIdx.x;
  if (idx >= Bc * SPc * Dc) return;
  int c = idx & 255, s = (idx >> 8) % SPc, b = (idx >> 8) / SPc;
  float v;
  if (s < Lc) {
    if (c < DEc) {
      v = h[((size_t)b * Lc + s) * DEc + c];
    } else {
      int j = c - DEc;
      float pos = (float)s * (1.0f / (Lc - 1));
      float acc = projb[j];
#pragma unroll
      for (int cc = 0; cc < 8; ++cc) {
        float f = (float)(1 << (cc >> 1));
        float ang = 6.28318530717958647f * f * pos;
        float e = (cc & 1) ? __cosf(ang) : __sinf(ang);
        acc += projw[j * 8 + cc] * e;
      }
      v = acc;
    }
  } else if (s == Lc) {
    v = stok[(size_t)b * Dc + c];
  } else {
    v = 0.0f;
  }
  hseq[idx] = v;
}

// ---------------- layernorm (256 cols, wave per row) ----------------
__global__ void k_ln(const float* __restrict__ in, const float* __restrict__ gm,
                     const float* __restrict__ bt, float* __restrict__ out, int rows) {
  int wave = threadIdx.x >> 5, lane = threadIdx.x & 31;
  int row = blockIdx.x * WAVES + wave;
  if (row >= rows) return;
  const float* x = in + (size_t)row * Dc;
  float v[8], s = 0.f, s2 = 0.f;
#pragma unroll
  for (int i = 0; i < 8; ++i) { v[i] = x[lane + 32 * i]; s += v[i]; s2 += v[i] * v[i]; }
#pragma unroll
  for (int off = 1; off < 32; off <<= 1) { s += __shfl_xor(s, off, 32); s2 += __shfl_xor(s2, off, 32); }
  float mean = s * (1.0f / Dc);
  float var  = s2 * (1.0f / Dc) - mean * mean;
  float rstd = rsqrtf(var + 1e-5f);
  float* o = out + (size_t)row * Dc;
#pragma unroll
  for (int i = 0; i < 8; ++i) { int c = lane + 32 * i; o[c] = (v[i] - mean) * rstd * gm[c] + bt[c]; }
}

// ---------------- LDS-tiled WMMA GEMM ----------------
// Block = 8 waves = 128 rows x 64 cols. B strip (64 x KCH bf16) staged in LDS via
// CDNA5 async copy (GLOBAL_LOAD_ASYNC_TO_LDS_B128 + s_wait_asynccnt), double-buffered.
// Each wave reuses one A-fragment across 4 N-tiles (4 f32 accumulators).
// Requires: rows % 128 == 0, N % 64 == 0, K % KCH == 0.
__global__ void __launch_bounds__(256)
k_gemm(const float* __restrict__ A, const __bf16* __restrict__ Bp,
       const float* __restrict__ bias, const float* __restrict__ resid,
       float* __restrict__ out, int rows, int N, int K, int act) {
  int tid  = threadIdx.x;
  int wave = tid >> 5, lane = tid & 31;
  int colStrips = N >> 6;
  int rb = blockIdx.x / colStrips;
  int nbase = (blockIdx.x % colStrips) << 6;
  int m = lane & 15, g = lane >> 4, nl = lane & 15;
  int rowTile = rb * 128 + wave * 16;
  const float* arow = A + (size_t)(rowTile + m) * K;

  __shared__ __align__(64) __bf16 Bs[2][64][KCH];

  // async-stage one 64 x KCH bf16 chunk of B into LDS (all 256 threads, 4x16B each)
  auto stage = [&](int buf, int kc) {
#pragma unroll
    for (int it = 0; it < 4; ++it) {
      int idx = tid + 256 * it;                 // 0..1023
      int nn = idx >> 4, ko = (idx & 15) << 3;  // 16B per load
      unsigned lds = (unsigned)(size_t)&Bs[buf][nn][ko];
      unsigned long long ga =
          (unsigned long long)(size_t)(Bp + (size_t)(nbase + nn) * K + (size_t)kc * KCH + ko);
      asm volatile("global_load_async_to_lds_b128 %0, %1, off"
                   :: "v"(lds), "v"(ga) : "memory");
    }
  };

  int nchunks = K / KCH;
  stage(0, 0);
  asm volatile("s_wait_asynccnt 0x0" ::: "memory");
  __syncthreads();

  v8f acc[4] = {};
  for (int c = 0; c < nchunks; ++c) {
    if (c + 1 < nchunks) stage((c + 1) & 1, c + 1);   // prefetch next chunk
    const __bf16* Bb = &Bs[c & 1][0][0];
#pragma unroll
    for (int kk = 0; kk < KCH; kk += 32) {
      v16bf a = a_frag_f32(arow + c * KCH + kk, g);
#pragma unroll
      for (int j = 0; j < 4; ++j) {
        v16bf bb = *(const v16bf*)(Bb + (size_t)(j * 16 + nl) * KCH + kk + 16 * g);
        acc[j] = WMMA_BF16(a, bb, acc[j]);
      }
    }
    asm volatile("s_wait_asynccnt 0x0" ::: "memory");
    __syncthreads();
  }

#pragma unroll
  for (int j = 0; j < 4; ++j) {
    int n = nbase + j * 16 + nl;
    float bv = bias ? bias[n] : 0.0f;
#pragma unroll
    for (int r = 0; r < 8; ++r) {
      int row = rowTile + r + 8 * g;
      float v = acc[j][r] + bv;
      if (act) v = gelu_exact(v);
      if (resid) v += resid[(size_t)row * N + n];
      out[(size_t)row * N + n] = v;
    }
  }
}

// ---------------- QKV scatter ----------------
__global__ void k_qkv_scatter(const float* __restrict__ qkv, __bf16* __restrict__ qb,
                              __bf16* __restrict__ kb, __bf16* __restrict__ vt) {
  int idx = blockIdx.x * blockDim.x + threadIdx.x;
  if (idx >= Rc * Dc) return;
  int j = idx & 255, row = idx >> 8;
  int s = row % SPc, b = row / SPc;
  int h = j >> 5, hd = j & 31;
  int bh = b * Hc + h;
  size_t base = (size_t)row * (3 * Dc);
  qb[((size_t)bh * SPc + s) * HDc + hd] = (__bf16)(qkv[base + j] * 0.17677669529663688f);
  kb[((size_t)bh * SPc + s) * HDc + hd] = (__bf16)qkv[base + Dc + j];
  vt[((size_t)bh * HDc + hd) * SPc + s] = (__bf16)qkv[base + 2 * Dc + j];
}

// ---------------- flash attention (diag-masked), one 16-query tile per wave ----------------
__global__ void k_attn(const __bf16* __restrict__ qb, const __bf16* __restrict__ kb,
                       const __bf16* __restrict__ vt, float* __restrict__ abuf) {
  int wave = threadIdx.x >> 5, lane = threadIdx.x & 31;
  int tid = blockIdx.x * WAVES + wave;
  const int QT = SPc / 16;                       // 65
  if (tid >= Bc * Hc * QT) return;
  int bh = tid / QT, qt = tid % QT;
  int b = bh >> 3, h = bh & 7;
  int m = lane & 15, g = lane >> 4, n = lane & 15;

  __shared__ __align__(64) __bf16 Psh[WAVES][16][32];
  __bf16 (*P)[32] = Psh[wave];

  v16bf qa = a_frag_bf16(qb + ((size_t)bh * SPc + qt * 16 + m) * HDc, g);
  v8f o0 = {}, o1 = {};
  float mr[8], lr[8];
#pragma unroll
  for (int r = 0; r < 8; ++r) { mr[r] = -1e30f; lr[r] = 0.0f; }

  for (int k0 = 0; k0 < SPc; k0 += 32) {
    const __bf16* krow = kb + ((size_t)bh * SPc + k0 + n) * HDc + 16 * g;
    v16bf bk0 = *(const v16bf*)krow;
    v16bf bk1 = *(const v16bf*)(krow + (size_t)16 * HDc);
    v8f s0 = {}, s1 = {};
    s0 = WMMA_BF16(qa, bk0, s0);
    s1 = WMMA_BF16(qa, bk1, s1);
#pragma unroll
    for (int r = 0; r < 8; ++r) {
      int sq = qt * 16 + r + 8 * g;
      float v0 = s0[r], v1 = s1[r];
      int key0 = k0 + n, key1 = k0 + 16 + n;
      if (key0 == sq || key0 >= Sc) v0 = -1e30f;
      if (key1 == sq || key1 >= Sc) v1 = -1e30f;
      float mx = fmaxf(v0, v1);
#pragma unroll
      for (int off = 1; off < 16; off <<= 1) mx = fmaxf(mx, __shfl_xor(mx, off, 32));
      float nm   = fmaxf(mr[r], mx);
      float corr = __expf(mr[r] - nm);
      float p0 = __expf(v0 - nm), p1 = __expf(v1 - nm);
      float ls = p0 + p1;
#pragma unroll
      for (int off = 1; off < 16; off <<= 1) ls += __shfl_xor(ls, off, 32);
      lr[r] = lr[r] * corr + ls;
      mr[r] = nm;
      o0[r] *= corr; o1[r] *= corr;
      P[r + 8 * g][n]      = (__bf16)p0;
      P[r + 8 * g][16 + n] = (__bf16)p1;
    }
    v16bf pa = a_frag_bf16(&P[m][0], g);
    const __bf16* vrow = vt + ((size_t)bh * HDc + n) * SPc + k0 + 16 * g;
    v16bf bv0 = *(const v16bf*)vrow;
    v16bf bv1 = *(const v16bf*)(vrow + (size_t)16 * SPc);
    o0 = WMMA_BF16(pa, bv0, o0);
    o1 = WMMA_BF16(pa, bv1, o1);
  }
#pragma unroll
  for (int r = 0; r < 8; ++r) {
    int s = qt * 16 + r + 8 * g;
    float invl = 1.0f / lr[r];
    size_t base = ((size_t)b * SPc + s) * Dc + h * HDc;
    abuf[base + n]      = o0[r] * invl;
    abuf[base + 16 + n] = o1[r] * invl;
  }
}

// ---------------- head: edge-padded conv(K=3, 256ch->1) + instance de-norm ----------------
__global__ void k_head(const float* __restrict__ xn, const float* __restrict__ hw,
                       const float* __restrict__ hb, const float* __restrict__ so,
                       float* __restrict__ out) {
  int idx = blockIdx.x * blockDim.x + threadIdx.x;
  if (idx >= Bc * Lc) return;
  int l = idx & (Lc - 1), b = idx >> 10;
  float acc = hb[0];
#pragma unroll
  for (int t = 0; t < 3; ++t) {
    int lp = l + t - 1;
    lp = lp < 0 ? 0 : (lp > Lc - 1 ? Lc - 1 : lp);
    const float* row = xn + ((size_t)b * SPc + lp) * Dc;
#pragma unroll 4
    for (int c = 0; c < Dc; ++c) acc += hw[c * 3 + t] * row[c];
  }
  out[idx] = acc * so[b * 2 + 1] + so[b * 2 + 0];
}

// ---------------- host launcher ----------------
extern "C" void kernel_launch(void* const* d_in, const int* in_sizes, int n_in,
                              void* d_out, int out_size, void* d_ws, size_t ws_size,
                              hipStream_t stream) {
  (void)in_sizes; (void)n_in; (void)out_size; (void)ws_size;
  const float* x = (const float*)d_in[0];
  const float *emb_w[4], *emb_b[4], *bng[4], *bnb[4], *bnrm[4], *bnrv[4];
  for (int i = 0; i < 4; ++i) {
    int o = 1 + 6 * i;
    emb_w[i] = (const float*)d_in[o];     emb_b[i] = (const float*)d_in[o + 1];
    bng[i]   = (const float*)d_in[o + 2]; bnb[i]   = (const float*)d_in[o + 3];
    bnrm[i]  = (const float*)d_in[o + 4]; bnrv[i]  = (const float*)d_in[o + 5];
  }
  const float* projw = (const float*)d_in[25]; const float* projb = (const float*)d_in[26];
  const float* ps1w  = (const float*)d_in[27]; const float* ps1b  = (const float*)d_in[28];
  const float* ps2w  = (const float*)d_in[29]; const float* ps2b  = (const float*)d_in[30];
  struct Lp { const float *ln1g,*ln1b,*wq,*wk,*wv,*wo,*ln2g,*ln2b,*w1,*b1,*w2,*b2; } Lw[3];
  for (int i = 0; i < 3; ++i) {
    int o = 31 + 12 * i;
    Lw[i] = { (const float*)d_in[o],      (const float*)d_in[o + 1], (const float*)d_in[o + 2],
              (const float*)d_in[o + 3],  (const float*)d_in[o + 4], (const float*)d_in[o + 5],
              (const float*)d_in[o + 6],  (const float*)d_in[o + 7], (const float*)d_in[o + 8],
              (const float*)d_in[o + 9],  (const float*)d_in[o + 10],(const float*)d_in[o + 11] };
  }
  const float* lnfg  = (const float*)d_in[67]; const float* lnfb  = (const float*)d_in[68];
  const float* headw = (const float*)d_in[69]; const float* headb = (const float*)d_in[70];

  // -------- workspace carve (256B aligned regions) --------
  char* p = (char*)d_ws;
  auto take = [&](size_t bytes) -> char* {
    char* r = p; p += (bytes + 255) & ~(size_t)255; return r;
  };
  float*  mn   = (float*)take((size_t)Bc * Lc * 4);
  float*  stok = (float*)take((size_t)Bc * Dc * 4);
  float*  so   = (float*)take((size_t)Bc * 2 * 4);
  float*  h0   = (float*)take((size_t)Bc * Lc * DEc * 4);
  float*  h1   = (float*)take((size_t)Bc * Lc * DEc * 4);
  float*  hseq = (float*)take((size_t)Rc * Dc * 4);
  float*  xn   = (float*)take((size_t)Rc * Dc * 4);
  float*  abuf = (float*)take((size_t)Rc * Dc * 4);
  float*  big  = (float*)take((size_t)Rc * FFc * 4);          // QKV fp32 and FFN hidden
  __bf16* qb   = (__bf16*)take((size_t)Bc * Hc * SPc * HDc * 2);
  __bf16* kb   = (__bf16*)take((size_t)Bc * Hc * SPc * HDc * 2);
  __bf16* vt   = (__bf16*)take((size_t)Bc * Hc * HDc * SPc * 2);
  __bf16* cpk  = (__bf16*)take((size_t)3 * 3 * DEc * DEc * 2);
  __bf16* qkvp[3]; __bf16* wop[3]; __bf16* w1p[3]; __bf16* w2p[3];
  for (int i = 0; i < 3; ++i) {
    qkvp[i] = (__bf16*)take((size_t)3 * Dc * Dc * 2);
    wop[i]  = (__bf16*)take((size_t)Dc * Dc * 2);
    w1p[i]  = (__bf16*)take((size_t)FFc * Dc * 2);
    w2p[i]  = (__bf16*)take((size_t)Dc * FFc * 2);
  }

  // -------- weight packing --------
  for (int i = 1; i < 4; ++i)
    k_pack_conv<<<(3 * DEc * DEc + 255) / 256, 256, 0, stream>>>(emb_w[i], cpk + (size_t)(i - 1) * 3 * DEc * DEc);
  for (int i = 0; i < 3; ++i) {
    k_pack<<<(Dc * Dc + 255) / 256, 256, 0, stream>>>(Lw[i].wq, qkvp[i],                     Dc, Dc);
    k_pack<<<(Dc * Dc + 255) / 256, 256, 0, stream>>>(Lw[i].wk, qkvp[i] + (size_t)Dc * Dc,   Dc, Dc);
    k_pack<<<(Dc * Dc + 255) / 256, 256, 0, stream>>>(Lw[i].wv, qkvp[i] + (size_t)2*Dc * Dc, Dc, Dc);
    k_pack<<<(Dc * Dc + 255) / 256, 256, 0, stream>>>(Lw[i].wo, wop[i], Dc, Dc);
    k_pack<<<(Dc * FFc + 255) / 256, 256, 0, stream>>>(Lw[i].w1, w1p[i], Dc, FFc);
    k_pack<<<(FFc * Dc + 255) / 256, 256, 0, stream>>>(Lw[i].w2, w2p[i], FFc, Dc);
  }

  // -------- embedding path --------
  k_instnorm<<<Bc, 256, 0, stream>>>(x, mn, ps1w, ps1b, ps2w, ps2b, stok, so);
  k_embed0<<<(Bc * Lc * DEc + 255) / 256, 256, 0, stream>>>(
      mn, emb_w[0], emb_b[0], bng[0], bnb[0], bnrm[0], bnrv[0], h0);
  const int convTiles = Bc * (Lc / 16) * (DEc / 16);
  const int convBlocks = (convTiles + WAVES - 1) / WAVES;
  k_conv<<<convBlocks, 32 * WAVES, 0, stream>>>(h0, cpk + 0 * (size_t)3 * DEc * DEc,
      emb_b[1], bng[1], bnb[1], bnrm[1], bnrv[1], h1, 2);
  k_conv<<<convBlocks, 32 * WAVES, 0, stream>>>(h1, cpk + 1 * (size_t)3 * DEc * DEc,
      emb_b[2], bng[2], bnb[2], bnrm[2], bnrv[2], h0, 4);
  k_conv<<<convBlocks, 32 * WAVES, 0, stream>>>(h0, cpk + 2 * (size_t)3 * DEc * DEc,
      emb_b[3], bng[3], bnb[3], bnrm[3], bnrv[3], h1, 8);
  k_build_hseq<<<(Bc * SPc * Dc + 255) / 256, 256, 0, stream>>>(h1, projw, projb, stok, hseq);

  // -------- transformer layers --------
  const int lnBlocks = (Rc + WAVES - 1) / WAVES;
  auto gemm_grid = [](int rows, int N) { return (rows / 128) * (N / 64); };
  for (int i = 0; i < 3; ++i) {
    k_ln<<<lnBlocks, 32 * WAVES, 0, stream>>>(hseq, Lw[i].ln1g, Lw[i].ln1b, xn, Rc);
    k_gemm<<<gemm_grid(Rc, 3 * Dc), 256, 0, stream>>>(
        xn, qkvp[i], nullptr, nullptr, big, Rc, 3 * Dc, Dc, 0);
    k_qkv_scatter<<<(Rc * Dc + 255) / 256, 256, 0, stream>>>(big, qb, kb, vt);
    k_attn<<<(Bc * Hc * (SPc / 16) + WAVES - 1) / WAVES, 32 * WAVES, 0, stream>>>(qb, kb, vt, abuf);
    k_gemm<<<gemm_grid(Rc, Dc), 256, 0, stream>>>(
        abuf, wop[i], nullptr, xn, hseq, Rc, Dc, Dc, 0);
    k_ln<<<lnBlocks, 32 * WAVES, 0, stream>>>(hseq, Lw[i].ln2g, Lw[i].ln2b, xn, Rc);
    k_gemm<<<gemm_grid(Rc, FFc), 256, 0, stream>>>(
        xn, w1p[i], Lw[i].b1, nullptr, big, Rc, FFc, Dc, 1);
    k_gemm<<<gemm_grid(Rc, Dc), 256, 0, stream>>>(
        big, w2p[i], Lw[i].b2, xn, hseq, Rc, Dc, FFc, 0);
  }

  // -------- final LN + head --------
  k_ln<<<lnBlocks, 32 * WAVES, 0, stream>>>(hseq, lnfg, lnfb, xn, Rc);
  k_head<<<(Bc * Lc + 255) / 256, 256, 0, stream>>>(xn, headw, headb, so, (float*)d_out);
}
